// RGCN_Net_17154099380785
// MI455X (gfx1250) — compile-verified, matrix operand order checked by reading, not run on hardware
//
#include <hip/hip_runtime.h>

typedef __attribute__((ext_vector_type(16))) _Float16 v16h;
typedef __attribute__((ext_vector_type(8)))  _Float16 v8h;
typedef __attribute__((ext_vector_type(4)))  _Float16 v4h;
typedef __attribute__((ext_vector_type(8)))  float    v8f;

// ---------------------------------------------------------------------------
// Elementwise helpers
// ---------------------------------------------------------------------------
__global__ void cast_f32_to_f16(const float* __restrict__ in,
                                _Float16* __restrict__ out, long n) {
  long i = (long)blockIdx.x * blockDim.x + threadIdx.x;
  long stride = (long)gridDim.x * blockDim.x;
  for (; i < n; i += stride) out[i] = (_Float16)in[i];
}

__global__ void zero_f32(float* __restrict__ p, long n) {
  long i = (long)blockIdx.x * blockDim.x + threadIdx.x;
  long stride = (long)gridDim.x * blockDim.x;
  for (; i < n; i += stride) p[i] = 0.0f;
}

__global__ void relu_cast_f16(const float* __restrict__ in,
                              _Float16* __restrict__ out, long n) {
  long i = (long)blockIdx.x * blockDim.x + threadIdx.x;
  long stride = (long)gridDim.x * blockDim.x;
  for (; i < n; i += stride) {
    float v = in[i];
    out[i] = (_Float16)(v > 0.0f ? v : 0.0f);
  }
}

// ---------------------------------------------------------------------------
// Pack f32 weights [Rcount x K x 64] directly into f16 WMMA B-fragments.
// Packed layout: frag(r, kt, nt)[lane][i] contiguous 16 halves per lane, so
// the GEMM loads each fragment as two coalesced global_load_b128.
//   kOff = (lane&16)?8:0 ; n = nt*16 + (lane&15)
//   i<8 : kk = kt*32 + kOff + i        ; i>=8 : kk = kt*32 + kOff + 16 + (i-8)
// ---------------------------------------------------------------------------
__global__ void pack_b_frags(const float* __restrict__ B,
                             _Float16* __restrict__ Bp, int K, long total) {
  long t = (long)blockIdx.x * blockDim.x + threadIdx.x;
  if (t >= total) return;
  int i    = (int)(t & 15);
  int lane = (int)((t >> 4) & 31);
  int nt   = (int)((t >> 9) & 3);
  long u   = t >> 11;
  int kTiles = K >> 5;
  int kt = (int)(u % kTiles);
  int r  = (int)(u / kTiles);
  int kOff = (lane & 16) ? 8 : 0;
  int n = nt * 16 + (lane & 15);
  int kk = kt * 32 + kOff + ((i < 8) ? i : (16 + (i - 8)));
  Bp[t] = (_Float16)B[((long)r * K + kk) * 64 + n];
}

// ---------------------------------------------------------------------------
// Per-(relation, dst) in-degree (computed once, reused by both layers)
// ---------------------------------------------------------------------------
__global__ void deg_count(const int* __restrict__ dst,
                          const int* __restrict__ etype,
                          float* __restrict__ deg, int E, int N) {
  int e = blockIdx.x * blockDim.x + threadIdx.x;
  if (e >= E) return;
  atomicAdd(&deg[(long)etype[e] * N + dst[e]], 1.0f);
}

// ---------------------------------------------------------------------------
// Edge scatter with mean normalization (the bandwidth roofline term):
//   agg[dst, :] += xw16[etype, src, :] / max(deg[etype, dst], 1)
// 16 lanes/edge, 8B f16 gather per lane (128B/edge), fp32 atomics into the
// L2-resident 25.6MB accumulator.
// ---------------------------------------------------------------------------
__global__ void scatter_mean(const _Float16* __restrict__ xw16,
                             const int* __restrict__ src,
                             const int* __restrict__ dst,
                             const int* __restrict__ etype,
                             const float* __restrict__ deg,
                             float* __restrict__ agg, int E, int N) {
  long g = (long)blockIdx.x * blockDim.x + threadIdx.x;
  long e = g >> 4;
  if (e >= E) return;
  int l = (int)(g & 15);
  int s = src[e];
  int d = dst[e];
  int r = etype[e];
  float dg = deg[(long)r * N + d];
  float inv = 1.0f / fmaxf(dg, 1.0f);
  v4h v = *(const v4h*)(xw16 + ((long)r * N + s) * 64 + l * 4);
  float* out = agg + (long)d * 64 + l * 4;
  atomicAdd(out + 0, (float)v[0] * inv);
  atomicAdd(out + 1, (float)v[1] * inv);
  atomicAdd(out + 2, (float)v[2] * inv);
  atomicAdd(out + 3, (float)v[3] * inv);
}

// ---------------------------------------------------------------------------
// WMMA GEMM: C[M x 64] = A[M x K] @ B[K x 64] + bias  (bias always valid)
// A f16 row-major; B pre-packed fragments; fp32 accumulation.
// Block = 128 threads = 4 waves; wave owns a 16x64 tile (4 v8f accumulators).
// blockIdx.y = relation slice.
// ---------------------------------------------------------------------------
template <bool OUT_F16>
__global__ __launch_bounds__(128) void wmma_gemm64(
    const _Float16* __restrict__ A, const _Float16* __restrict__ Bpack,
    void* __restrict__ Cout, const float* __restrict__ bias, int M, int K) {
  const int tid  = threadIdx.x;
  const int wave = tid >> 5;
  const int lane = tid & 31;
  const int mBase = (blockIdx.x * 4 + wave) * 16;
  const int kTiles = K >> 5;

  int rowA = mBase + (lane & 15);
  if (rowA >= M) rowA = M - 1;  // clamp; tail stores guarded below
  const int kLaneOff = (lane & 16) ? 8 : 0;
  const int nCol = lane & 15;

  // Per-relation packed-B base; fragment stride between n-tiles = 512 halves.
  const _Float16* bp =
      Bpack + ((long)blockIdx.y * kTiles * 4) * 512 + (long)lane * 16;

  v8f acc[4] = {};

  for (int kt = 0; kt < kTiles; ++kt) {
    // A fragment: two contiguous 16B chunks per lane
    const _Float16* pa = A + (long)rowA * K + kt * 32 + kLaneOff;
    v8h alo = *(const v8h*)(pa);
    v8h ahi = *(const v8h*)(pa + 16);
    v16h a;
#pragma unroll
    for (int i = 0; i < 8; ++i) { a[i] = alo[i]; a[8 + i] = ahi[i]; }

    const _Float16* bk = bp + (long)kt * 4 * 512;
#pragma unroll
    for (int nt = 0; nt < 4; ++nt) {
      v16h b = *(const v16h*)(bk + nt * 512);
      acc[nt] = __builtin_amdgcn_wmma_f32_16x16x32_f16(false, a, false, b,
                                                       (short)0, acc[nt],
                                                       false, false);
    }
  }

  // --- Epilogue ---
  const int mOff = (lane & 16) ? 8 : 0;
  float bv0 = bias[nCol];
  float bv1 = bias[16 + nCol];
  float bv2 = bias[32 + nCol];
  float bv3 = bias[48 + nCol];
  float bv[4] = {bv0, bv1, bv2, bv3};
  float*    Cf = (float*)Cout    + (long)blockIdx.y * M * 64;
  _Float16* Ch = (_Float16*)Cout + (long)blockIdx.y * M * 64;

  if (mBase + 16 <= M) {  // full tile: unguarded stores
#pragma unroll
    for (int nt = 0; nt < 4; ++nt) {
      long base = (long)(mBase + mOff) * 64 + nt * 16 + nCol;
#pragma unroll
      for (int v = 0; v < 8; ++v) {
        float val = acc[nt][v] + bv[nt];
        if (OUT_F16) Ch[base + (long)v * 64] = (_Float16)val;
        else         Cf[base + (long)v * 64] = val;
      }
    }
  } else {  // tail tile only (last M-tile)
#pragma unroll
    for (int nt = 0; nt < 4; ++nt) {
      const int col = nt * 16 + nCol;
#pragma unroll
      for (int v = 0; v < 8; ++v) {
        int row = mBase + mOff + v;
        if (row < M) {
          float val = acc[nt][v] + bv[nt];
          if (OUT_F16) Ch[(long)row * 64 + col] = (_Float16)val;
          else         Cf[(long)row * 64 + col] = val;
        }
      }
    }
  }
}

// ---------------------------------------------------------------------------
// Launch
// ---------------------------------------------------------------------------
extern "C" void kernel_launch(void* const* d_in, const int* in_sizes, int n_in,
                              void* d_out, int out_size, void* d_ws, size_t ws_size,
                              hipStream_t stream) {
  const float* x     = (const float*)d_in[0];
  const int*   eidx  = (const int*)d_in[1];
  const int*   etype = (const int*)d_in[2];
  const float* W1    = (const float*)d_in[3];
  const float* b1    = (const float*)d_in[5];
  const float* root1 = (const float*)d_in[4];
  const float* W2    = (const float*)d_in[6];
  const float* root2 = (const float*)d_in[7];
  const float* b2    = (const float*)d_in[8];

  const int C_IN = 128, HID = 64;
  const int N = in_sizes[0] / C_IN;
  const int E = in_sizes[2];
  const int R = in_sizes[3] / (C_IN * HID);
  const int* src = eidx;
  const int* dst = eidx + E;

  // --- carve workspace ---
  char* ws = (char*)d_ws;
  size_t off = 0;
  auto carve = [&](size_t bytes) -> void* {
    void* p = ws + off;
    off = (off + bytes + 255) & ~(size_t)255;
    return p;
  };
  _Float16* x16  = (_Float16*)carve((size_t)N * C_IN * 2);      // 25.6 MB
  _Float16* w1p  = (_Float16*)carve((size_t)R * C_IN * HID * 2);// packed frags
  _Float16* r1p  = (_Float16*)carve((size_t)C_IN * HID * 2);
  _Float16* w2p  = (_Float16*)carve((size_t)R * HID * HID * 2);
  _Float16* r2p  = (_Float16*)carve((size_t)HID * HID * 2);
  float*    zb   = (float*)carve(64 * 4);                       // zero bias
  float*    deg  = (float*)carve((size_t)R * N * 4);            // 3.2 MB
  _Float16* xw   = (_Float16*)carve((size_t)R * N * HID * 2);   // 102.4 MB
  float*    agg  = (float*)carve((size_t)N * HID * 4);          // 25.6 MB
  _Float16* h16  = (_Float16*)carve((size_t)N * HID * 2);       // 12.8 MB

  const int T = 256;
  auto blocks = [](long n, int t) { return (int)((n + t - 1) / t); };

  // --- casts + weight fragment packing (fused cast+pack) ---
  cast_f32_to_f16<<<blocks((long)N * C_IN, T), T, 0, stream>>>(x, x16, (long)N * C_IN);
  {
    long t1 = (long)R * C_IN * HID;
    pack_b_frags<<<blocks(t1, T), T, 0, stream>>>(W1, w1p, C_IN, t1);
    long t2 = (long)C_IN * HID;
    pack_b_frags<<<blocks(t2, T), T, 0, stream>>>(root1, r1p, C_IN, t2);
    long t3 = (long)R * HID * HID;
    pack_b_frags<<<blocks(t3, T), T, 0, stream>>>(W2, w2p, HID, t3);
    long t4 = (long)HID * HID;
    pack_b_frags<<<blocks(t4, T), T, 0, stream>>>(root2, r2p, HID, t4);
  }
  zero_f32<<<1, 64, 0, stream>>>(zb, 64);

  // --- degrees (shared by both layers) ---
  zero_f32<<<blocks((long)R * N, T), T, 0, stream>>>(deg, (long)R * N);
  deg_count<<<blocks(E, T), T, 0, stream>>>(dst, etype, deg, E, N);

  const dim3 gemmBlk(128);
  const int mTiles = (N + 63) / 64;

  // --- layer 1 ---
  wmma_gemm64<true><<<dim3(mTiles, R), gemmBlk, 0, stream>>>(
      x16, w1p, (void*)xw, zb, N, C_IN);                       // xW[r] in f16
  wmma_gemm64<false><<<dim3(mTiles, 1), gemmBlk, 0, stream>>>(
      x16, r1p, (void*)agg, b1, N, C_IN);                      // agg = x@root1 + b1
  scatter_mean<<<blocks((long)E * 16, T), T, 0, stream>>>(
      xw, src, dst, etype, deg, agg, E, N);
  relu_cast_f16<<<blocks((long)N * HID, T), T, 0, stream>>>(agg, h16, (long)N * HID);

  // --- layer 2 (reuse xw buffer and deg) ---
  wmma_gemm64<true><<<dim3(mTiles, R), gemmBlk, 0, stream>>>(
      h16, w2p, (void*)xw, zb, N, HID);                        // hW[r] in f16
  wmma_gemm64<false><<<dim3(mTiles, 1), gemmBlk, 0, stream>>>(
      h16, r2p, d_out, b2, N, HID);                            // out = h@root2 + b2
  scatter_mean<<<blocks((long)E * 16, T), T, 0, stream>>>(
      xw, src, dst, etype, deg, (float*)d_out, E, N);
}
